// Decoder_65120294142587
// MI455X (gfx1250) — matrix operand order repaired and checked by reading, not compiled, not run on hardware
//
#include <hip/hip_runtime.h>
#include <hip/hip_bf16.h>

typedef __attribute__((ext_vector_type(16))) _Float16 v16h;
typedef __attribute__((ext_vector_type(8)))  _Float16 v8h;
typedef __attribute__((ext_vector_type(8)))  float    v8f;

#define CDIV(a,b) (((a)+(b)-1)/(b))

// ---------------------------------------------------------------------------
// Core WMMA GEMM:  C[M x N] (f32, ldc) = A[M x K](f16, lda) @ W[N x K](f16, ldw)^T + bias[n]
// M multiple of 16, N multiple of 64 (padded buffers); K multiple of 32.
// One wave32 computes FOUR consecutive 16x16 n-tiles, sharing a single A-tile
// fetch per k-step: 10 b128 loads feed 4 v_wmma_f32_16x16x32_f16 (vs 4 loads/1 wmma).
// grid = (ceil(N/256), M/16), block = 128 (4 waves; wave w owns n-group blockIdx.x*4+w)
// ---------------------------------------------------------------------------
__global__ __launch_bounds__(128)
void k_gemm_f16(const _Float16* __restrict__ A, const _Float16* __restrict__ W,
                const float* __restrict__ bias, float* __restrict__ C,
                int M, int N, int K, int lda, int ldw, int ldc)
{
    int wave = threadIdx.x >> 5;
    int lane = threadIdx.x & 31;
    int g  = blockIdx.x * 4 + wave;           // group of 4 consecutive n-tiles
    int mt = blockIdx.y;
    if (g * 64 >= N) return;                  // wave-uniform exit (EXEC all-1 for WMMA)
    int l16 = lane & 15;
    int hi  = lane >> 4;
    // A tile (16x32 f16): lane<16 -> row l16, K {0..15}; lane>=16 -> K {16..31}
    const _Float16* arow = A + (size_t)(mt * 16 + l16) * lda + hi * 16;
    // B tiles (32x16 f16): lane holds column l16, K-range hi*16.. -> contiguous W row bytes
    const _Float16* wrow = W + (size_t)(g * 64 + l16) * ldw + hi * 16;
    size_t wstep = (size_t)16 * ldw;

    v8f c0 = {0.f,0.f,0.f,0.f,0.f,0.f,0.f,0.f};
    v8f c1 = c0, c2 = c0, c3 = c0;
    for (int k0 = 0; k0 < K; k0 += 32) {
        v8h a0 = *(const v8h*)(arow + k0);
        v8h a1 = *(const v8h*)(arow + k0 + 8);
        v16h a;
#pragma unroll
        for (int i = 0; i < 8; ++i) { a[i] = a0[i]; a[8 + i] = a1[i]; }
        v16h b0 = *(const v16h*)(wrow + k0);
        v16h b1 = *(const v16h*)(wrow + wstep + k0);
        v16h b2 = *(const v16h*)(wrow + 2 * wstep + k0);
        v16h b3 = *(const v16h*)(wrow + 3 * wstep + k0);
        c0 = __builtin_amdgcn_wmma_f32_16x16x32_f16(false, a, false, b0, (short)0, c0, false, false);
        c1 = __builtin_amdgcn_wmma_f32_16x16x32_f16(false, a, false, b1, (short)0, c1, false, false);
        c2 = __builtin_amdgcn_wmma_f32_16x16x32_f16(false, a, false, b2, (short)0, c2, false, false);
        c3 = __builtin_amdgcn_wmma_f32_16x16x32_f16(false, a, false, b3, (short)0, c3, false, false);
    }
    v8f accs[4] = {c0, c1, c2, c3};
#pragma unroll
    for (int j = 0; j < 4; ++j) {
        int col = g * 64 + j * 16 + l16;
        float bb = bias ? bias[col] : 0.f;
#pragma unroll
        for (int i = 0; i < 8; ++i) {
            int m = mt * 16 + hi * 8 + i;      // C layout: VGPR i -> row hi*8+i, col l16
            C[(size_t)m * ldc + col] = accs[j][i] + bb;
        }
    }
}

// ---------------------------------------------------------------------------
// Init helpers
// ---------------------------------------------------------------------------
__global__ void k_f32_to_f16(const float* __restrict__ s, _Float16* __restrict__ d, int n)
{
    int i = blockIdx.x * blockDim.x + threadIdx.x;
    if (i < n) d[i] = (_Float16)s[i];
}

__global__ void k_zero(float* p, int n)
{
    int i = blockIdx.x * blockDim.x + threadIdx.x;
    if (i < n) p[i] = 0.f;
}

// enc (S,B,256) f32 -> enc_h (b*256+s, 256) f16   (row-major A for the wenc GEMM)
__global__ void k_enc_to_f16(const float* __restrict__ enc, _Float16* __restrict__ d)
{
    int i = blockIdx.x * blockDim.x + threadIdx.x;
    if (i >= 2048 * 256) return;
    int dd = i & 255;
    int s  = (i >> 8) & 255;
    int b  = i >> 16;
    d[i] = (_Float16)enc[((size_t)s * 8 + b) * 256 + dd];
}

// combined spec/stop weight: W (128 x 1280) f16 (rows 0..79 spec, 80 stop, rest 0) + bias(128)
__global__ void k_build_ss(const float* __restrict__ specw, const float* __restrict__ stopw,
                           const float* __restrict__ specb, const float* __restrict__ stopb,
                           _Float16* __restrict__ W, float* __restrict__ bias)
{
    int i = blockIdx.x * blockDim.x + threadIdx.x;
    if (i < 128 * 1280) {
        int r = i / 1280, k = i % 1280;
        float v = 0.f;
        if (r < 80) v = specw[r * 1280 + k];
        else if (r == 80) v = stopw[k];
        W[i] = (_Float16)v;
    }
    if (i < 128) bias[i] = (i < 80) ? specb[i] : (i == 80 ? stopb[0] : 0.f);
}

// ---------------------------------------------------------------------------
// Per-step small kernels
// ---------------------------------------------------------------------------
__global__ void k_prenet(const float* __restrict__ prev,
                         const float* __restrict__ fc1w, const float* __restrict__ fc1b,
                         const float* __restrict__ fc2w, const float* __restrict__ fc2b,
                         float* __restrict__ q)
{
    __shared__ float q1[8][256];
    int j = threadIdx.x;   // 256 threads
    for (int b = 0; b < 8; ++b) {
        float s = fc1b[j];
        for (int k = 0; k < 80; ++k) s += prev[b * 80 + k] * fc1w[j * 80 + k];
        q1[b][j] = fmaxf(s, 0.f);
    }
    __syncthreads();
    for (int b = 0; b < 8; ++b) {
        float s = fc2b[j];
        for (int k = 0; k < 256; ++k) s += q1[b][k] * fc2w[j * 256 + k];
        q[b * 256 + j] = fmaxf(s, 0.f);
    }
}

// energies + softmax; writes attn (8x256) and masks[b,s,t] when t>=0
__global__ void k_attn(const float* __restrict__ wenc, const float* __restrict__ q,
                       float* __restrict__ attn, float* __restrict__ mask_base, int t)
{
    __shared__ float red[256];
    int b = blockIdx.x, s = threadIdx.x;
    const float* wr = wenc + ((size_t)b * 256 + s) * 256;
    const float* qb = q + b * 256;
    float acc = 0.f;
    for (int d = 0; d < 256; ++d) acc += wr[d] * qb[d];
    red[s] = acc; __syncthreads();
    for (int off = 128; off > 0; off >>= 1) {
        if (s < off) red[s] = fmaxf(red[s], red[s + off]);
        __syncthreads();
    }
    float mx = red[0]; __syncthreads();
    float ex = __expf(acc - mx);
    red[s] = ex; __syncthreads();
    for (int off = 128; off > 0; off >>= 1) {
        if (s < off) red[s] += red[s + off];
        __syncthreads();
    }
    float a = ex / red[0];
    attn[b * 256 + s] = a;
    if (t >= 0) mask_base[(size_t)b * 25600 + s * 100 + t] = a;
}

// context[b,d] = sum_s enc[s,b,d] * attn[b,s]
__global__ void k_context(const float* __restrict__ enc, const float* __restrict__ attn,
                          float* __restrict__ ctx)
{
    __shared__ float al[256];
    int b = blockIdx.x, d = threadIdx.x;
    al[d] = attn[b * 256 + d];
    __syncthreads();
    float acc = 0.f;
    for (int s = 0; s < 256; ++s) acc += enc[((size_t)s * 8 + b) * 256 + d] * al[s];
    ctx[b * 256 + d] = acc;
}

// stage [a(8xKa) | b(8xKb)] -> f16 dst (16 x (Ka+Kb)), rows 8..15 zero-padded
__global__ void k_stage2(const float* __restrict__ a, int Ka,
                         const float* __restrict__ b_, int Kb, _Float16* __restrict__ dst)
{
    int K = Ka + Kb;
    int i = blockIdx.x * blockDim.x + threadIdx.x;
    if (i >= 16 * K) return;
    int r = i / K, k = i % K;
    float v = 0.f;
    if (r < 8) v = (k < Ka) ? a[r * Ka + k] : b_[r * Kb + (k - Ka)];
    dst[i] = (_Float16)v;
}

// GRU gates with h_prev = 0:  h = (1-z)*n ; gx already contains x@Wih^T + bih
__global__ void k_gru_gates(const float* __restrict__ gx, const float* __restrict__ bhh,
                            float* __restrict__ h)
{
    int i = blockIdx.x * blockDim.x + threadIdx.x;
    if (i >= 8 * 1024) return;
    int b = i >> 10, j = i & 1023;
    const float* g = gx + (size_t)b * 3072;
    float r = 1.f / (1.f + __expf(-(g[j] + bhh[j])));
    float z = 1.f / (1.f + __expf(-(g[1024 + j] + bhh[1024 + j])));
    float n = tanhf(g[2048 + j] + r * bhh[2048 + j]);
    h[i] = (1.f - z) * n;
}

__global__ void k_stop(const float* __restrict__ ss, float* __restrict__ out, int t)
{
    if (t < 0) return;
    int b = threadIdx.x;
    if (b < 8) out[64000 + b * 100 + t] = ss[b * 128 + 80];
}

// postnet layer1: conv(1->512,k5,pad2) + BN + tanh ; y layout (b, co, l)
__global__ void k_postnet1(const float* __restrict__ ss, const float* __restrict__ w,
                           const float* __restrict__ bias, const float* __restrict__ g,
                           const float* __restrict__ be, const float* __restrict__ m,
                           const float* __restrict__ v, float* __restrict__ y)
{
    int i = blockIdx.x * blockDim.x + threadIdx.x;
    if (i >= 8 * 512 * 80) return;
    int l = i % 80, co = (i / 80) & 511, b = i / 40960;
    const float* sp = ss + b * 128;
    float acc = bias[co];
#pragma unroll
    for (int dk = 0; dk < 5; ++dk) {
        int lp = l + dk - 2;
        if (lp >= 0 && lp < 80) acc += w[co * 5 + dk] * sp[lp];
    }
    float sc = g[co] * rsqrtf(v[co] + 1e-5f);
    y[i] = tanhf((acc - m[co]) * sc + be[co]);
}

// im2col: col[(b*80+l), ci*5+dk] = y[b,ci,l+dk-2]  (f16, zero pad)
__global__ void k_im2col(const float* __restrict__ y, _Float16* __restrict__ col)
{
    int i = blockIdx.x * blockDim.x + threadIdx.x;
    if (i >= 640 * 2560) return;
    int kk = i % 2560, mrow = i / 2560;
    int b = mrow / 80, l = mrow % 80;
    int ci = kk / 5, dk = kk % 5;
    int lp = l + dk - 2;
    float vv = (lp >= 0 && lp < 80) ? y[(size_t)b * 40960 + ci * 80 + lp] : 0.f;
    col[i] = (_Float16)vv;
}

// BN + tanh epilogue for conv GEMM output Cc (640 x 512, row = b*80+l, col = co)
__global__ void k_bntanh(const float* __restrict__ Cc, const float* __restrict__ g,
                         const float* __restrict__ be, const float* __restrict__ m,
                         const float* __restrict__ v, float* __restrict__ y)
{
    int i = blockIdx.x * blockDim.x + threadIdx.x;
    if (i >= 8 * 512 * 80) return;
    int l = i % 80, co = (i / 80) & 511, b = i / 40960;
    float x = Cc[(size_t)(b * 80 + l) * 512 + co];
    float sc = g[co] * rsqrtf(v[co] + 1e-5f);
    y[i] = tanhf((x - m[co]) * sc + be[co]);
}

// layer5 conv(512->1) + residual; updates carry `prev`, writes outs[t] when t>=0
__global__ void k_postnet5(const float* __restrict__ y, const float* __restrict__ w5,
                           const float* __restrict__ b5, const float* __restrict__ ss,
                           float* __restrict__ prev, float* __restrict__ out, int t)
{
    __shared__ float wl[2560];
    for (int k = threadIdx.x; k < 2560; k += blockDim.x) wl[k] = w5[k];
    __syncthreads();
    int i = blockIdx.x * blockDim.x + threadIdx.x;
    if (i >= 640) return;
    int b = i / 80, l = i % 80;
    float acc = b5[0];
    for (int ci = 0; ci < 512; ++ci) {
        const float* yc = y + (size_t)b * 40960 + ci * 80;
        const float* wc = wl + ci * 5;
#pragma unroll
        for (int dk = 0; dk < 5; ++dk) {
            int lp = l + dk - 2;
            if (lp >= 0 && lp < 80) acc += wc[dk] * yc[lp];
        }
    }
    float spec = ss[b * 128 + l] + acc;
    prev[i] = spec;
    if (t >= 0) out[(size_t)t * 640 + i] = spec;
}

// ---------------------------------------------------------------------------
// Host driver
// ---------------------------------------------------------------------------
extern "C" void kernel_launch(void* const* d_in, const int* in_sizes, int n_in,
                              void* d_out, int out_size, void* d_ws, size_t ws_size,
                              hipStream_t stream)
{
    (void)in_sizes; (void)n_in; (void)out_size; (void)ws_size;
    // setup_inputs() flat order (params dict in insertion order):
    const float* enc   = (const float*)d_in[0];
    const float* fc1w  = (const float*)d_in[2];
    const float* fc1b  = (const float*)d_in[3];
    const float* fc2w  = (const float*)d_in[4];
    const float* fc2b  = (const float*)d_in[5];
    const float* attnW = (const float*)d_in[6];
    const float* wih0  = (const float*)d_in[7];
    const float* bih0  = (const float*)d_in[9];
    const float* bhh0  = (const float*)d_in[10];
    const float* wih1  = (const float*)d_in[11];
    const float* bih1  = (const float*)d_in[13];
    const float* bhh1  = (const float*)d_in[14];
    const float* specw = (const float*)d_in[15];
    const float* specb = (const float*)d_in[16];
    const float* stopw = (const float*)d_in[17];
    const float* stopb = (const float*)d_in[18];
    const float* pn5w  = (const float*)d_in[19];
    const float* pn5b  = (const float*)d_in[20];
    const float* pn1w  = (const float*)d_in[21];
    const float* pn1b  = (const float*)d_in[22];
    const float* pn1g  = (const float*)d_in[23];
    const float* pn1be = (const float*)d_in[24];
    const float* pn1m  = (const float*)d_in[25];
    const float* pn1v  = (const float*)d_in[26];
    const float* pnw[3]  = {(const float*)d_in[27], (const float*)d_in[33], (const float*)d_in[39]};
    const float* pnb[3]  = {(const float*)d_in[28], (const float*)d_in[34], (const float*)d_in[40]};
    const float* png[3]  = {(const float*)d_in[29], (const float*)d_in[35], (const float*)d_in[41]};
    const float* pnbe[3] = {(const float*)d_in[30], (const float*)d_in[36], (const float*)d_in[42]};
    const float* pnm[3]  = {(const float*)d_in[31], (const float*)d_in[37], (const float*)d_in[43]};
    const float* pnv[3]  = {(const float*)d_in[32], (const float*)d_in[38], (const float*)d_in[44]};

    float* outF = (float*)d_out;
    float* maskB = outF + 64800;   // outs(64000) + stops(800)

    // --- workspace carve-up ---
    char* base = (char*)d_ws;
    size_t off = 0;
    auto carve = [&](size_t bytes) -> char* {
        char* p = base + off;
        off += (bytes + 255) & ~(size_t)255;
        return p;
    };
    _Float16* enc_h   = (_Float16*)carve(2048 * 256 * 2);
    _Float16* attnW_h = (_Float16*)carve(256 * 256 * 2);
    _Float16* wih0_h  = (_Float16*)carve(3072 * 512 * 2);
    _Float16* wih1_h  = (_Float16*)carve(3072 * 1024 * 2);
    _Float16* wss_h   = (_Float16*)carve(128 * 1280 * 2);
    float*    bias_ss = (float*)carve(128 * 4);
    _Float16* wpn_h[3];
    for (int i = 0; i < 3; ++i) wpn_h[i] = (_Float16*)carve(512 * 2560 * 2);
    float* wenc  = (float*)carve(2048 * 256 * 4);
    float* prev  = (float*)carve(640 * 4);
    float* q     = (float*)carve(2048 * 4);
    float* attn  = (float*)carve(2048 * 4);
    float* ctx   = (float*)carve(2048 * 4);
    _Float16* x_h    = (_Float16*)carve(16 * 512 * 2);
    float*    gx     = (float*)carve(16 * 3072 * 4);
    float*    h1     = (float*)carve(8 * 1024 * 4);
    float*    h2     = (float*)carve(8 * 1024 * 4);
    _Float16* h1_h   = (_Float16*)carve(16 * 1024 * 2);
    _Float16* feat_h = (_Float16*)carve(16 * 1280 * 2);
    float*    ssb    = (float*)carve(16 * 128 * 4);
    float*    y0     = (float*)carve(8 * 512 * 80 * 4);
    float*    y1     = (float*)carve(8 * 512 * 80 * 4);
    _Float16* col_h  = (_Float16*)carve(640 * 2560 * 2);
    float*    convC  = (float*)carve(640 * 512 * 4);

    // --- one-time (per launch) prep: f16 weight copies, combined spec/stop, wenc ---
    k_enc_to_f16<<<CDIV(2048 * 256, 256), 256, 0, stream>>>(enc, enc_h);
    k_f32_to_f16<<<CDIV(65536, 256), 256, 0, stream>>>(attnW, attnW_h, 65536);
    k_f32_to_f16<<<CDIV(1572864, 256), 256, 0, stream>>>(wih0, wih0_h, 1572864);
    k_f32_to_f16<<<CDIV(3145728, 256), 256, 0, stream>>>(wih1, wih1_h, 3145728);
    for (int i = 0; i < 3; ++i)
        k_f32_to_f16<<<CDIV(1310720, 256), 256, 0, stream>>>(pnw[i], wpn_h[i], 1310720);
    k_build_ss<<<CDIV(128 * 1280, 256), 256, 0, stream>>>(specw, stopw, specb, stopb, wss_h, bias_ss);
    k_zero<<<CDIV(640, 256), 256, 0, stream>>>(prev, 640);
    // wenc = enc_bsd @ attn_W^T : M=2048, N=256, K=256
    k_gemm_f16<<<dim3(1, 128), 128, 0, stream>>>(enc_h, attnW_h, nullptr, wenc,
                                                 2048, 256, 256, 256, 256, 256);

    // --- sequential decode: call 0 builds the carry; calls 1..100 emit outputs ---
    for (int c = 0; c <= 100; ++c) {
        int t = c - 1;
        k_prenet<<<1, 256, 0, stream>>>(prev, fc1w, fc1b, fc2w, fc2b, q);
        k_attn<<<8, 256, 0, stream>>>(wenc, q, attn, maskB, t);
        k_context<<<8, 256, 0, stream>>>(enc, attn, ctx);
        // x = [q | ctx] padded to 16 rows, f16
        k_stage2<<<CDIV(16 * 512, 256), 256, 0, stream>>>(q, 256, ctx, 256, x_h);
        // GRU0: gx = x @ wih0^T + bih0   (M=16, N=3072, K=512)
        k_gemm_f16<<<dim3(12, 1), 128, 0, stream>>>(x_h, wih0_h, bih0, gx,
                                                    16, 3072, 512, 512, 512, 3072);
        k_gru_gates<<<CDIV(8192, 256), 256, 0, stream>>>(gx, bhh0, h1);
        k_stage2<<<CDIV(16 * 1024, 256), 256, 0, stream>>>(h1, 1024, h1, 0, h1_h);
        // GRU1: gx = h1 @ wih1^T + bih1  (M=16, N=3072, K=1024)
        k_gemm_f16<<<dim3(12, 1), 128, 0, stream>>>(h1_h, wih1_h, bih1, gx,
                                                    16, 3072, 1024, 1024, 1024, 3072);
        k_gru_gates<<<CDIV(8192, 256), 256, 0, stream>>>(gx, bhh1, h2);
        // feat = [h2 | ctx]
        k_stage2<<<CDIV(16 * 1280, 256), 256, 0, stream>>>(h2, 1024, ctx, 256, feat_h);
        // spec/stop: ss = feat @ Wss^T + bias  (M=16, N=128 padded, K=1280)
        k_gemm_f16<<<dim3(1, 1), 128, 0, stream>>>(feat_h, wss_h, bias_ss, ssb,
                                                   16, 128, 1280, 1280, 1280, 128);
        k_stop<<<1, 32, 0, stream>>>(ssb, outF, t);
        // postnet
        k_postnet1<<<CDIV(327680, 256), 256, 0, stream>>>(ssb, pn1w, pn1b, pn1g, pn1be,
                                                          pn1m, pn1v, y0);
        float* ysrc = y0; float* ydst = y1;
        for (int L = 0; L < 3; ++L) {   // layers 2,3,4 as WMMA GEMMs (640x512x2560)
            k_im2col<<<CDIV(640 * 2560, 256), 256, 0, stream>>>(ysrc, col_h);
            k_gemm_f16<<<dim3(2, 40), 128, 0, stream>>>(col_h, wpn_h[L], pnb[L], convC,
                                                        640, 512, 2560, 2560, 2560, 512);
            k_bntanh<<<CDIV(327680, 256), 256, 0, stream>>>(convC, png[L], pnbe[L],
                                                            pnm[L], pnv[L], ydst);
            float* tmp = ysrc; ysrc = ydst; ydst = tmp;
        }
        // after 3 ping-pong swaps ysrc holds layer-4 output
        k_postnet5<<<CDIV(640, 256), 256, 0, stream>>>(ysrc, pn5w, pn5b, ssb, prev, outF, t);
    }
}
